// NeighborhoodAttention_10316511445126
// MI455X (gfx1250) — matrix operand order repaired and checked by reading, compiled, not run on hardware
//
#include <hip/hip_runtime.h>

// ---- problem constants (match reference) ----
#define KS   7
#define NS   3
#define NHD  4
#define DIM  128
#define HD   32          // DIM / NHD
#define BB   4
#define HH   56
#define WW   56
#define KK   128         // GEMM reduction dim (both GEMMs)

typedef _Float16 half_t;
typedef __attribute__((ext_vector_type(16))) _Float16 v16h;
typedef __attribute__((ext_vector_type(8)))  _Float16 v8h;
typedef __attribute__((ext_vector_type(8)))  float    v8f;
typedef __attribute__((ext_vector_type(4)))  unsigned int u32x4;
typedef __attribute__((ext_vector_type(8)))  int      i32x8;
typedef __attribute__((ext_vector_type(4)))  int      i32x4;

// ---------------------------------------------------------------------------
// elementwise f32 -> f16
// ---------------------------------------------------------------------------
__global__ __launch_bounds__(256)
void cvt_f32_to_f16(const float* __restrict__ in, half_t* __restrict__ out, int n)
{
    int i = blockIdx.x * blockDim.x + threadIdx.x;
    if (i < n) out[i] = (half_t)in[i];
}

// w[K][N] (f32) -> wT[N][K] (f16)
__global__ __launch_bounds__(256)
void transpose_to_f16(const float* __restrict__ w, half_t* __restrict__ wT, int K, int N)
{
    int i = blockIdx.x * blockDim.x + threadIdx.x;   // i = n*K + k
    if (i < K * N) {
        int k = i % K, n = i / K;
        wT[i] = (half_t)w[(size_t)k * N + n];
    }
}

// ---------------------------------------------------------------------------
// GEMM: C[M,N] = A[M,128] @ BT[N,128]^T + bias[N]   (f16 in, f32 out)
// grid = (M/64, N/64), block = 128 threads (4 waves), one 64x64 tile/block.
// A panel (64x128 f16 = 16KB) staged into LDS by the Tensor Data Mover;
// B fragments are two 16B runs (32B apart) in the transposed weight -> b128 loads.
// ---------------------------------------------------------------------------
__global__ __launch_bounds__(128)
void gemm_wmma_f16(const half_t* __restrict__ A,    // [M][128] row-major
                   const half_t* __restrict__ BT,   // [N][128] row-major (B transposed)
                   const float* __restrict__ bias,
                   float* __restrict__ C,
                   int M, int N)
{
    __shared__ half_t Apanel[64][KK];                // 16 KB

    const int lane = threadIdx.x & 31;
    const int wv   = threadIdx.x >> 5;
    const int mblk = blockIdx.x * 64;
    const int n0   = blockIdx.y * 64;

#if __has_builtin(__builtin_amdgcn_tensor_load_to_lds) && __has_builtin(__builtin_amdgcn_s_wait_tensorcnt)
    if (wv == 0) {
        // ---- D# descriptor (CDNA5 ISA): 2-D tile ----
        const unsigned long long gaddr =
            (unsigned long long)(const void*)(A + (size_t)mblk * KK);
        const unsigned int lds_base =
            (unsigned int)(unsigned long long)&Apanel[0][0];   // LDS offset = low 32 bits

        u32x4 g0;
        g0[0] = 1u;                                   // count=1, user descriptor
        g0[1] = lds_base;                             // lds_addr
        g0[2] = (unsigned int)gaddr;                  // global_addr[31:0]
        g0[3] = ((unsigned int)(gaddr >> 32) & 0x01FFFFFFu) | 0x80000000u; // addr[56:32] | type=2

        i32x8 g1;
        g1[0] = 0x00010000;          // workgroup_mask=0, data_size=1 (2 bytes)
        g1[1] = (KK & 0xFFFF) << 16; // tensor_dim0[15:0] = 128
        g1[2] = (64 << 16);          // tensor_dim0[31:16]=0 | tensor_dim1[15:0]=64
        g1[3] = (KK << 16);          // tensor_dim1[31:16]=0 | tile_dim0=128
        g1[4] = 64;                  // tile_dim1=64, tile_dim2=0
        g1[5] = KK;                  // tensor_dim0_stride[31:0] = 128 elements
        g1[6] = 0;                   // stride0[47:32]=0, dim1_stride[15:0]=0
        g1[7] = 0;

        i32x4 gz = {0, 0, 0, 0};
#if __clang_major__ >= 23
        i32x8 gz8 = {0, 0, 0, 0, 0, 0, 0, 0};
        __builtin_amdgcn_tensor_load_to_lds(g0, g1, gz, gz, gz8, 0);
#else
        __builtin_amdgcn_tensor_load_to_lds(g0, g1, gz, gz, 0);
#endif
        __builtin_amdgcn_s_wait_tensorcnt(0);
    }
#else
    // fallback: cooperative vector copy of the A panel
    {
        const v8h* src = (const v8h*)(A + (size_t)mblk * KK);
        v8h* dst = (v8h*)&Apanel[0][0];
        for (int i = threadIdx.x; i < 64 * KK / 8; i += 128)
            dst[i] = src[i];
    }
#endif
    __syncthreads();

    const int m_row = (lane & 15) + wv * 16;   // A row for this lane's fragment
    const int kb    = (lane >> 4) * 8;         // A frag K sub-offset (ISA A 16x32 layout)
    const int nn    = lane & 15;               // B/D column within 16-tile
    const int kB    = (lane >> 4) * 8;         // B frag K sub-offset (same k order as A)

    const v8f vzero = {0.f,0.f,0.f,0.f,0.f,0.f,0.f,0.f};
    v8f acc[4];
#pragma unroll
    for (int t = 0; t < 4; ++t) acc[t] = vzero;

#pragma unroll
    for (int k0 = 0; k0 < KK; k0 += 32) {
        const v8h alo = *(const v8h*)&Apanel[m_row][k0 + kb];
        const v8h ahi = *(const v8h*)&Apanel[m_row][k0 + kb + 16];
        const v16h afrag = __builtin_shufflevector(alo, ahi,
            0,1,2,3,4,5,6,7,8,9,10,11,12,13,14,15);
#pragma unroll
        for (int t = 0; t < 4; ++t) {
            const half_t* bp = BT + (size_t)(n0 + t * 16 + nn) * KK + k0 + kB;
            const v8h blo = *(const v8h*)bp;
            const v8h bhi = *(const v8h*)(bp + 16);
            const v16h bfrag = __builtin_shufflevector(blo, bhi,
                0,1,2,3,4,5,6,7,8,9,10,11,12,13,14,15);
            acc[t] = __builtin_amdgcn_wmma_f32_16x16x32_f16(
                false, afrag, false, bfrag, (short)0, acc[t], false, false);
        }
    }

    // ---- store D (VGPR r: M = r + 8*(lane>=16), N = lane&15) ----
    {
        const int mh = (lane >> 4) * 8;
#pragma unroll
        for (int t = 0; t < 4; ++t) {
            const int col = n0 + t * 16 + nn;
            const float bv = bias[col];
#pragma unroll
            for (int r = 0; r < 8; ++r)
                C[(size_t)(mblk + wv * 16 + mh + r) * N + col] = acc[t][r] + bv;
        }
    }
}

// ---------------------------------------------------------------------------
// Neighborhood attention core: one thread per (b, head, h, w).
// qkv layout per token: [q(128) | k(128) | v(128)] (f32).
// Output f16, layout out[token*128 + head*32 + d] (A operand of proj GEMM).
// ---------------------------------------------------------------------------
__global__ __launch_bounds__(256)
void natten_kernel(const float* __restrict__ qkv,
                   const float* __restrict__ rpb,
                   half_t* __restrict__ out)
{
    const int tid   = blockIdx.x * blockDim.x + threadIdx.x;
    const int total = BB * NHD * HH * WW;
    if (tid >= total) return;

    int w = tid % WW;
    int t = tid / WW;
    int h = t % HH;  t /= HH;
    int head = t % NHD;
    int b = t / NHD;

    const float scale = 0.17677669529663687f;   // 32^-0.5

    int hs = h - NS; if (hs < 0) hs = 0; if (hs > HH - KS) hs = HH - KS;
    int ws = w - NS; if (ws < 0) ws = 0; if (ws > WW - KS) ws = WW - KS;
    const int ph = (h < NS) ? (KS - 1 - h) : ((h + NS >= HH) ? (HH - h - 1) : NS);
    const int pw = (w < NS) ? (KS - 1 - w) : ((w + NS >= WW) ? (WW - w - 1) : NS);

    const size_t tok = ((size_t)(b * HH + h) * WW + w);
    const float* qp = qkv + tok * (3 * DIM) + head * HD;

    float q[HD];
#pragma unroll
    for (int d = 0; d < HD; ++d) q[d] = qp[d] * scale;

    float sc[KS * KS];
    float mx = -1e30f;
#pragma unroll
    for (int kh = 0; kh < KS; ++kh) {
#pragma unroll
        for (int kw = 0; kw < KS; ++kw) {
            const size_t ntok = ((size_t)(b * HH + (hs + kh)) * WW + (ws + kw));
            const float* kp = qkv + ntok * (3 * DIM) + DIM + head * HD;
            float s = 0.f;
#pragma unroll
            for (int d = 0; d < HD; ++d) s += q[d] * kp[d];
            s += rpb[((size_t)head * (2 * KS - 1) + (ph + kh)) * (2 * KS - 1) + (pw + kw)];
            sc[kh * KS + kw] = s;
            mx = fmaxf(mx, s);
        }
    }

    float den = 0.f;
#pragma unroll
    for (int i = 0; i < KS * KS; ++i) {
        sc[i] = __expf(sc[i] - mx);
        den += sc[i];
    }
    const float inv = 1.f / den;

    float o[HD];
#pragma unroll
    for (int d = 0; d < HD; ++d) o[d] = 0.f;

#pragma unroll
    for (int kh = 0; kh < KS; ++kh) {
#pragma unroll
        for (int kw = 0; kw < KS; ++kw) {
            const size_t ntok = ((size_t)(b * HH + (hs + kh)) * WW + (ws + kw));
            const float* vp = qkv + ntok * (3 * DIM) + 2 * DIM + head * HD;
            const float p = sc[kh * KS + kw] * inv;
#pragma unroll
            for (int d = 0; d < HD; ++d) o[d] += p * vp[d];
        }
    }

    half_t* op = out + tok * DIM + head * HD;
#pragma unroll
    for (int d = 0; d < HD; ++d) op[d] = (half_t)o[d];
}

// ---------------------------------------------------------------------------
extern "C" void kernel_launch(void* const* d_in, const int* in_sizes, int n_in,
                              void* d_out, int out_size, void* d_ws, size_t ws_size,
                              hipStream_t stream)
{
    const float* x      = (const float*)d_in[0];
    const float* w_qkv  = (const float*)d_in[1];
    const float* b_qkv  = (const float*)d_in[2];
    const float* rpb    = (const float*)d_in[3];
    const float* w_proj = (const float*)d_in[4];
    const float* b_proj = (const float*)d_in[5];
    float* out = (float*)d_out;

    const int M = BB * HH * WW;          // 12544 tokens (divisible by 64)

    // ---- workspace layout (f16 slabs first, then f32 qkv) ----
    half_t* xh     = (half_t*)d_ws;                       // M*128
    half_t* wqkvT  = xh + (size_t)M * DIM;                // 384*128
    half_t* wprojT = wqkvT + (size_t)(3 * DIM) * DIM;     // 128*128
    half_t* attn_h = wprojT + (size_t)DIM * DIM;          // M*128
    float*  qkv_ws = (float*)(attn_h + (size_t)M * DIM);  // M*384

    // 0) precision staging: x -> f16, weights -> f16 transposed [N][K]
    {
        const int n = M * DIM;
        cvt_f32_to_f16<<<(n + 255) / 256, 256, 0, stream>>>(x, xh, n);
    }
    transpose_to_f16<<<(DIM * 3 * DIM + 255) / 256, 256, 0, stream>>>(
        w_qkv, wqkvT, DIM, 3 * DIM);
    transpose_to_f16<<<(DIM * DIM + 255) / 256, 256, 0, stream>>>(
        w_proj, wprojT, DIM, DIM);

    // 1) QKV projection: xh(M,128) @ w_qkv(128,384) + b_qkv -> f32
    gemm_wmma_f16<<<dim3(M / 64, (3 * DIM) / 64), 128, 0, stream>>>(
        xh, wqkvT, b_qkv, qkv_ws, M, 3 * DIM);

    // 2) windowed attention (per token, per head) -> f16
    natten_kernel<<<(M * NHD + 255) / 256, 256, 0, stream>>>(
        qkv_ws, rpb, attn_h);

    // 3) output projection: attn(M,128) @ w_proj(128,128) + b_proj -> f32
    gemm_wmma_f16<<<dim3(M / 64, DIM / 64), 128, 0, stream>>>(
        attn_h, wprojT, b_proj, out, M, DIM);
}
